// SelfAttention_35931696398380
// MI455X (gfx1250) — matrix operand order repaired and checked by reading, not compiled, not run on hardware
//
#include <hip/hip_runtime.h>

#define BB 8
#define HH 8
#define SS 2048
#define DD 64
#define NEGV -1e9f
// 1/sqrt(64) * log2(e): scores leave WMMA already in base-2 domain, so each
// softmax exp is a single v_exp_f32 (hardware exp2).
#define QSCALE 0.18033688011112042f

typedef __attribute__((ext_vector_type(16))) _Float16 v16h;
typedef __attribute__((ext_vector_type(8)))  _Float16 v8h;
typedef __attribute__((ext_vector_type(4)))  _Float16 v4h;
typedef __attribute__((ext_vector_type(8)))  float    v8f;
typedef __attribute__((ext_vector_type(2)))  float    v2f;

#if __has_builtin(__builtin_amdgcn_tensor_load_to_lds) && __has_builtin(__builtin_amdgcn_s_wait_tensorcnt)
#define HAVE_TDM 1
typedef __attribute__((ext_vector_type(4))) unsigned int u32x4;
typedef __attribute__((ext_vector_type(8))) int          i32x8;
typedef __attribute__((ext_vector_type(4))) int          i32x4;
#else
#define HAVE_TDM 0
#endif

union H8 { v8h h; int i[4]; };

// ---- VOP3P packed-f32 helpers: 2 lanes of math per VALU issue slot ----
static __device__ inline v2f pk_add(v2f a, v2f b) {
    v2f d; asm("v_pk_add_f32 %0, %1, %2" : "=v"(d) : "v"(a), "v"(b)); return d;
}
static __device__ inline v2f pk_mul(v2f a, v2f b) {
    v2f d; asm("v_pk_mul_f32 %0, %1, %2" : "=v"(d) : "v"(a), "v"(b)); return d;
}

static __device__ inline v8h shflx16(v8h v) {
    H8 a; a.h = v; H8 r;
#pragma unroll
    for (int k = 0; k < 4; ++k) r.i[k] = __shfl_xor(a.i[k], 16, 32);
    return r.h;
}

static __device__ inline v16h cat8(v8h lo, v8h hi) {
    return __builtin_shufflevector(lo, hi, 0,1,2,3,4,5,6,7,8,9,10,11,12,13,14,15);
}

static __device__ inline v16h cvt16(float4 a, float4 b, float4 c, float4 d, float s) {
    v16h r;
    r[0]=(_Float16)(a.x*s);  r[1]=(_Float16)(a.y*s);  r[2]=(_Float16)(a.z*s);  r[3]=(_Float16)(a.w*s);
    r[4]=(_Float16)(b.x*s);  r[5]=(_Float16)(b.y*s);  r[6]=(_Float16)(b.z*s);  r[7]=(_Float16)(b.w*s);
    r[8]=(_Float16)(c.x*s);  r[9]=(_Float16)(c.y*s);  r[10]=(_Float16)(c.z*s); r[11]=(_Float16)(c.w*s);
    r[12]=(_Float16)(d.x*s); r[13]=(_Float16)(d.y*s); r[14]=(_Float16)(d.z*s); r[15]=(_Float16)(d.w*s);
    return r;
}

static __device__ inline v16h ldB(const float* p, float s) {
    const float4* f = (const float4*)p;
    return cvt16(f[0], f[1], f[2], f[3], s);
}

static __device__ inline v16h ldsA(const _Float16* row) {
    return cat8(*(const v8h*)row, *(const v8h*)(row + 8));
}

#if HAVE_TDM
// TDM DMA of a 64x64 f32 tile (row stride 64 elems) global -> LDS.
// group1 (dims/strides) is loop-invariant and passed in pre-built.
static __device__ inline void tdm_issue(unsigned lds_off, unsigned long long ga, i32x8 g1) {
    u32x4 g0;
    g0[0] = 1u;                                                // count=1, user mode
    g0[1] = lds_off;                                           // LDS byte address
    g0[2] = (unsigned)(ga & 0xFFFFFFFFu);                      // global_addr[31:0]
    g0[3] = (unsigned)((ga >> 32) & 0x01FFFFFFu) | (2u << 30); // addr[56:32] | type=2
    i32x4 gz = {0, 0, 0, 0};
#if defined(__clang_major__) && (__clang_major__ >= 23)
    i32x8 gz8 = {0, 0, 0, 0, 0, 0, 0, 0};
    __builtin_amdgcn_tensor_load_to_lds(g0, g1, gz, gz, gz8, 0);
#else
    __builtin_amdgcn_tensor_load_to_lds(g0, g1, gz, gz, 0);
#endif
}
#endif

__global__ __launch_bounds__(256)
void fa_fwd_f16wmma(const float* __restrict__ Q, const float* __restrict__ K,
                    const float* __restrict__ V, const int* __restrict__ Msk,
                    float* __restrict__ Out) {
    __shared__ __align__(16) _Float16 Kt[64][72];    // row-major K tile (f16)
    __shared__ __align__(16) _Float16 Vt[64][72];    // V^T tile (f16)
#if HAVE_TDM
    __shared__ __align__(16) float rawK[2][64 * 64]; // TDM double buffers (raw f32)
    __shared__ __align__(16) float rawV[2][64 * 64];
#endif

    const int tid  = threadIdx.x;
    const int lane = tid & 31;
    const int wave = tid >> 5;
    const int ql   = lane & 15;      // query index on lanes (N of S^T / O^T)
    const int hi   = lane >> 4;      // half-wave

    const int hIdx = blockIdx.y;
    const int bIdx = blockIdx.z;
    const int bh   = bIdx * HH + hIdx;
    const int q0   = blockIdx.x * 128 + wave * 16;

    const float* Qb = Q + (size_t)bh * SS * DD;
    const float* Kb = K + (size_t)bh * SS * DD;
    const float* Vb = V + (size_t)bh * SS * DD;
    const int*   Mb = Msk + ((size_t)bIdx * SS + (size_t)(q0 + ql)) * SS;

    // Q^T B-fragments, pre-scaled by 1/sqrt(D)*log2(e) (loaded once)
    const float* qrow = Qb + (size_t)(q0 + ql) * DD;
    const v16h bQ0 = ldB(qrow + 16 * hi,      QSCALE);
    const v16h bQ1 = ldB(qrow + 32 + 16 * hi, QSCALE);

    v8f acc[4];
#pragma unroll
    for (int dt = 0; dt < 4; ++dt)
#pragma unroll
        for (int e = 0; e < 8; ++e) acc[dt][e] = 0.0f;

    float m_q = -__builtin_inff();
    float l_q = 0.0f;

    const int sr = tid >> 4;          // 0..15 : staging row within quad
    const int sc = (tid & 15) * 4;    // staging column base

#if HAVE_TDM
    const unsigned ldsK[2] = { (unsigned)(__UINTPTR_TYPE__)&rawK[0][0],
                               (unsigned)(__UINTPTR_TYPE__)&rawK[1][0] };
    const unsigned ldsV[2] = { (unsigned)(__UINTPTR_TYPE__)&rawV[0][0],
                               (unsigned)(__UINTPTR_TYPE__)&rawV[1][0] };
    i32x8 g1t;                         // loop-invariant D# group1
    g1t[0] = 0x20000;                  // data_size=2 (4B)
    g1t[1] = 64 << 16;                 // tensor_dim0 = 64
    g1t[2] = 64 << 16;                 // tensor_dim1 = 64
    g1t[3] = 64 << 16;                 // tile_dim0 = 64
    g1t[4] = 64;                       // tile_dim1 = 64
    g1t[5] = 64;                       // tensor_dim0_stride = 64
    g1t[6] = 0;
    g1t[7] = 0;
    unsigned long long gaK = (unsigned long long)(__UINTPTR_TYPE__)Kb;
    unsigned long long gaV = (unsigned long long)(__UINTPTR_TYPE__)Vb;
    const unsigned long long tileBytes = (unsigned long long)64 * DD * 4;
    if (wave == 0) {                   // kick off block 0 DMA
        tdm_issue(ldsK[0], gaK, g1t);
        tdm_issue(ldsV[0], gaV, g1t);
    }
#endif

    for (int j = 0; j < SS / 64; ++j) {
        const int kv0 = j * 64;

#if HAVE_TDM
        const int buf = j & 1;
        if (wave == 0) __builtin_amdgcn_s_wait_tensorcnt(0);  // block j's tiles landed
        __syncthreads();               // tiles visible; prior Kt/Vt reads done
        gaK += tileBytes;
        gaV += tileBytes;
        if ((j + 1 < SS / 64) && wave == 0) {                 // DMA block j+1 async
            tdm_issue(ldsK[buf ^ 1], gaK, g1t);
            tdm_issue(ldsV[buf ^ 1], gaV, g1t);
        }
        // ---- convert raw f32 tiles -> Kt (row-major f16) / Vt (transposed f16) ----
#pragma unroll
        for (int i = 0; i < 4; ++i) {
            const int row = sr + i * 16;
            float4 k4 = *(const float4*)&rawK[buf][row * 64 + sc];
            float4 u4 = *(const float4*)&rawV[buf][row * 64 + sc];
            v4h kh;
            kh[0] = (_Float16)k4.x; kh[1] = (_Float16)k4.y;
            kh[2] = (_Float16)k4.z; kh[3] = (_Float16)k4.w;
            *(v4h*)&Kt[row][sc] = kh;
            Vt[sc + 0][row] = (_Float16)u4.x;
            Vt[sc + 1][row] = (_Float16)u4.y;
            Vt[sc + 2][row] = (_Float16)u4.z;
            Vt[sc + 3][row] = (_Float16)u4.w;
        }
#else
        __syncthreads();               // previous iteration's LDS reads done
#pragma unroll
        for (int i = 0; i < 4; ++i) {
            const int row = sr + i * 16;
            float4 k4 = *(const float4*)(Kb + (size_t)(kv0 + row) * DD + sc);
            float4 u4 = *(const float4*)(Vb + (size_t)(kv0 + row) * DD + sc);
            v4h kh;
            kh[0] = (_Float16)k4.x; kh[1] = (_Float16)k4.y;
            kh[2] = (_Float16)k4.z; kh[3] = (_Float16)k4.w;
            *(v4h*)&Kt[row][sc] = kh;
            Vt[sc + 0][row] = (_Float16)u4.x;
            Vt[sc + 1][row] = (_Float16)u4.y;
            Vt[sc + 2][row] = (_Float16)u4.z;
            Vt[sc + 3][row] = (_Float16)u4.w;
        }
        if (kv0 + 64 < SS) {
            const float* pk = Kb + (size_t)(kv0 + 64 + (tid >> 2)) * DD + (tid & 3) * 16;
            const float* pv = Vb + (size_t)(kv0 + 64 + (tid >> 2)) * DD + (tid & 3) * 16;
            __builtin_prefetch(pk, 0, 3);
            __builtin_prefetch(pv, 0, 3);
        }
#endif
        if (kv0 + 64 < SS) {           // mask lines for next block
            __builtin_prefetch(Mb + kv0 + 64, 0, 3);
            __builtin_prefetch(Mb + kv0 + 96, 0, 3);
        }
        __syncthreads();               // Kt/Vt ready

        // ---- S^T = K_tile x Q^T  (4 tiles of 16 kv rows) + mask ----
        v8f st[4];
#pragma unroll
        for (int t = 0; t < 4; ++t) {
            const _Float16* krow = &Kt[16 * t + ql][0];
            v16h aK0 = ldsA(krow + 16 * hi);
            v16h aK1 = ldsA(krow + 32 + 16 * hi);
            v8f c;
#pragma unroll
            for (int e = 0; e < 8; ++e) c[e] = 0.0f;
            c = __builtin_amdgcn_wmma_f32_16x16x32_f16(false, aK0, false, bQ0, (short)0, c, false, false);
            c = __builtin_amdgcn_wmma_f32_16x16x32_f16(false, aK1, false, bQ1, (short)0, c, false, false);
            const int* mp = Mb + kv0 + 16 * t + 8 * hi;
            int4 m0 = *(const int4*)mp;
            int4 m1 = *(const int4*)(mp + 4);
            const int mm[8] = {m0.x, m0.y, m0.z, m0.w, m1.x, m1.y, m1.z, m1.w};
#pragma unroll
            for (int e = 0; e < 8; ++e) c[e] = mm[e] ? c[e] : NEGV;
            st[t] = c;
        }

        // ---- online softmax in base-2 domain (packed-f32 math) ----
        float rmax = -3.0e38f;
#pragma unroll
        for (int t = 0; t < 4; ++t)
#pragma unroll
            for (int e = 0; e < 8; ++e) rmax = fmaxf(rmax, st[t][e]);
        rmax = fmaxf(rmax, __shfl_xor(rmax, 16, 32));
        const float m_new = fmaxf(m_q, rmax);
        const float alpha = __builtin_amdgcn_exp2f(m_q - m_new);

        const v2f nm2 = { -m_new, -m_new };
        v2f sum2 = { 0.0f, 0.0f };
        H8 pt[4];
#pragma unroll
        for (int t = 0; t < 4; ++t)
#pragma unroll
            for (int i = 0; i < 4; ++i) {
                v2f x = { st[t][2 * i], st[t][2 * i + 1] };
                x = pk_add(x, nm2);                     // 2 subs / instruction
                v2f p;
                p[0] = __builtin_amdgcn_exp2f(x[0]);
                p[1] = __builtin_amdgcn_exp2f(x[1]);
                sum2 = pk_add(sum2, p);                 // 2 adds / instruction
                pt[t].h[2 * i]     = (_Float16)p[0];    // -> v_cvt_pk_f16_f32
                pt[t].h[2 * i + 1] = (_Float16)p[1];
            }
        float rsum = sum2[0] + sum2[1];
        rsum += __shfl_xor(rsum, 16, 32);
        l_q = l_q * alpha + rsum;
        m_q = m_new;

        const v2f al2 = { alpha, alpha };
#pragma unroll
        for (int dt = 0; dt < 4; ++dt)
#pragma unroll
            for (int i = 0; i < 4; ++i) {
                v2f a = { acc[dt][2 * i], acc[dt][2 * i + 1] };
                a = pk_mul(a, al2);                     // 2 muls / instruction
                acc[dt][2 * i]     = a[0];
                acc[dt][2 * i + 1] = a[1];
            }

        // ---- build P^T B-fragments via half-wave shuffles ----
        v8h s0 = shflx16(pt[0].h), s1 = shflx16(pt[1].h);
        v8h s2 = shflx16(pt[2].h), s3 = shflx16(pt[3].h);
        const bool lo = (hi == 0);
        v16h bP0 = cat8(lo ? pt[0].h : s1, lo ? s0 : pt[1].h);
        v16h bP1 = cat8(lo ? pt[2].h : s3, lo ? s2 : pt[3].h);

        // ---- O^T += V^T x P^T ----
#pragma unroll
        for (int dt = 0; dt < 4; ++dt) {
            const _Float16* vrow = &Vt[16 * dt + ql][0];
            v16h aV0 = ldsA(vrow + 16 * hi);
            v16h aV1 = ldsA(vrow + 32 + 16 * hi);
            acc[dt] = __builtin_amdgcn_wmma_f32_16x16x32_f16(false, aV0, false, bP0, (short)0, acc[dt], false, false);
            acc[dt] = __builtin_amdgcn_wmma_f32_16x16x32_f16(false, aV1, false, bP1, (short)0, acc[dt], false, false);
        }
    }

    // ---- epilogue: divide by l (packed), contiguous 2x b128 stores per d-tile ----
    const float inv = 1.0f / l_q;
    const v2f inv2 = { inv, inv };
    float* orow = Out + ((size_t)bh * SS + (size_t)(q0 + ql)) * DD;
#pragma unroll
    for (int dt = 0; dt < 4; ++dt) {
        float o[8];
#pragma unroll
        for (int i = 0; i < 4; ++i) {
            v2f a = { acc[dt][2 * i], acc[dt][2 * i + 1] };
            a = pk_mul(a, inv2);
            o[2 * i]     = a[0];
            o[2 * i + 1] = a[1];
        }
        float4 o0, o1;
        o0.x = o[0]; o0.y = o[1]; o0.z = o[2]; o0.w = o[3];
        o1.x = o[4]; o1.y = o[5]; o1.z = o[6]; o1.w = o[7];
        *(float4*)(orow + 16 * dt + 8 * hi)     = o0;
        *(float4*)(orow + 16 * dt + 8 * hi + 4) = o1;
    }
}

extern "C" void kernel_launch(void* const* d_in, const int* in_sizes, int n_in,
                              void* d_out, int out_size, void* d_ws, size_t ws_size,
                              hipStream_t stream) {
    (void)in_sizes; (void)n_in; (void)out_size; (void)d_ws; (void)ws_size;
    const float* Q = (const float*)d_in[0];
    const float* K = (const float*)d_in[1];
    const float* V = (const float*)d_in[2];
    const int*   M = (const int*)d_in[3];
    float* O = (float*)d_out;
    dim3 grid(SS / 128, HH, BB);   // b outermost: all heads of a batch share the mask slab in L2
    fa_fwd_f16wmma<<<grid, dim3(256), 0, stream>>>(Q, K, V, M, O);
}
